// SelfClassifyingHead_68616397520937
// MI455X (gfx1250) — matrix operand order repaired and checked by reading, not compiled, not run on hardware
//
#include <hip/hip_runtime.h>
#include <hip/hip_bf16.h>
#include <math.h>

#define BATCH    2048
#define DIM      1024
#define CLASSES  1024
#define OUT_DIM  1024
#define DEPTH    4
#define SEQ_LEN  8
#define HEADS    2
#define DIM_HEAD 64
#define INNER    128            // HEADS*DIM_HEAD
#define FF_INNER 2048
#define FF2      4096           // 2*FF_INNER
#define ROT_DIM  32

typedef __attribute__((ext_vector_type(16))) __bf16 v16bf;
typedef __attribute__((ext_vector_type(8)))  float  v8f;

// ---------------------------------------------------------------------------
// Weight conversion: f32 (K x N) row-major  ->  bf16 (N x K) row-major
// ---------------------------------------------------------------------------
__global__ void convert_transpose_kernel(const float* __restrict__ src,
                                         __bf16* __restrict__ dst,
                                         int K, int N, int rowOff, int dstStride) {
    size_t t = (size_t)blockIdx.x * blockDim.x + threadIdx.x;
    size_t total = (size_t)K * N;
    if (t >= total) return;
    int k = (int)(t / N);
    int n = (int)(t % N);
    dst[(size_t)(rowOff + n) * dstStride + k] = (__bf16)src[t];
}

__global__ void convert_kernel(const float* __restrict__ src,
                               __bf16* __restrict__ dst, size_t n) {
    size_t t = (size_t)blockIdx.x * blockDim.x + threadIdx.x;
    if (t < n) dst[t] = (__bf16)src[t];
}

__global__ void zero_u32_kernel(unsigned int* __restrict__ p, size_t n) {
    size_t t = (size_t)blockIdx.x * blockDim.x + threadIdx.x;
    if (t < n) p[t] = 0u;
}

__global__ void copy_kernel(const float* __restrict__ src, float* __restrict__ dst, size_t n) {
    size_t t = (size_t)blockIdx.x * blockDim.x + threadIdx.x;
    if (t < n) dst[t] = src[t];
}

// ---------------------------------------------------------------------------
// WMMA GEMM:  C[M x N] = A[M x K](f32) @ Wt[N x K](bf16)^T  (+residual/bias)
//   mode 0: C = acc
//   mode 1: C = acc + res[m][n]   (residual; res may alias C)
//   mode 2: C = acc + res[n]      (bias)
//
// Block = 256 threads = 8 waves. Block tile M=64 x N=128.
// Waves arranged 4(M) x 2(N); each wave computes a 16x64 tile as four
// independent 16x16 WMMA accumulators (4 wmma per K-step, independent
// accumulation chains hide the XDL RAW latency).
// A-tile (64x32 bf16) is double-buffered in LDS: stage t+1 while computing t,
// so only one workgroup barrier per 32-wide K-step.
// ---------------------------------------------------------------------------
__global__ __launch_bounds__(256) void gemm_bf16_wmma(
    const float* __restrict__ A, const __bf16* __restrict__ Wt,
    float* __restrict__ C, const float* __restrict__ res,
    int M, int N, int K, int mode)
{
    __shared__ __align__(16) __bf16 As[2][64][32];
    const int lane = threadIdx.x & 31;
    const int wave = threadIdx.x >> 5;            // 0..7
    const int wm   = wave >> 1;                   // 0..3  (M sub-tile)
    const int wn   = wave & 1;                    // 0..1  (N half)
    const int m0 = blockIdx.y * 64;
    const int n0 = blockIdx.x * 128 + wn * 64;
    const int row = lane & 15;                    // A row / W column within tile
    const int hi  = lane >> 4;                    // 0 or 1 (lane half)
    const int kh  = hi * 8;

    // staging assignment: thread stages 8 consecutive bf16 of one A row
    const int srow = threadIdx.x >> 2;            // 0..63
    const int scol = (threadIdx.x & 3) * 8;       // 0,8,16,24

    v8f acc0 = {}, acc1 = {}, acc2 = {}, acc3 = {};

    // stage A[m0+srow][k0+scol .. +7]  ->  As[buf][srow][scol..]
    auto stage = [&](int k0, int buf) {
        float f[8];
        __builtin_memcpy(f, A + (size_t)(m0 + srow) * K + k0 + scol, 32);
        __bf16 t[8];
        #pragma unroll
        for (int j = 0; j < 8; ++j) t[j] = (__bf16)f[j];
        __builtin_memcpy(&As[buf][srow][scol], t, 16);   // ds_store_b128
    };

    stage(0, 0);
    __syncthreads();

    const int nsteps = K / 32;
    for (int t = 0; t < nsteps; ++t) {
        const int buf = t & 1;
        const int k0  = t * 32;

        // A fragment: lane<16 -> K 0..15; lane>=16 -> K 16..31
        v16bf a;
        {
            const __bf16* ar = &As[buf][wm * 16 + row][0];
            __builtin_memcpy(&a, ar + hi * 16, 16);                      // ds_load_b128
            __builtin_memcpy(((__bf16*)&a) + 8, ar + hi * 16 + 8, 16);  // ds_load_b128
        }

        // B fragments: lane<16 -> K 0..15 of col (n0+j*16+row); lane>=16 -> K 16..31
        const __bf16* wp = Wt + (size_t)(n0 + row) * K + k0 + hi * 16;
        v16bf b0 = *(const v16bf*)(wp);
        v16bf b1 = *(const v16bf*)(wp + (size_t)16 * K);
        v16bf b2 = *(const v16bf*)(wp + (size_t)32 * K);
        v16bf b3 = *(const v16bf*)(wp + (size_t)48 * K);
        if (k0 + 32 < K) __builtin_prefetch(wp + 32, 0, 3);         // next K-slice

        // stage next tile into the other buffer while WMMAs execute
        if (t + 1 < nsteps) stage(k0 + 32, buf ^ 1);

        acc0 = __builtin_amdgcn_wmma_f32_16x16x32_bf16(false, a, false, b0, (short)0, acc0, false, false);
        acc1 = __builtin_amdgcn_wmma_f32_16x16x32_bf16(false, a, false, b1, (short)0, acc1, false, false);
        acc2 = __builtin_amdgcn_wmma_f32_16x16x32_bf16(false, a, false, b2, (short)0, acc2, false, false);
        acc3 = __builtin_amdgcn_wmma_f32_16x16x32_bf16(false, a, false, b3, (short)0, acc3, false, false);

        __syncthreads();   // one barrier per K-step (double-buffered LDS)
    }

    // C layout: VGPR r -> M = (m0 + wm*16) + 8*hi + r, N = n0 + j*16 + (lane&15)
    const int cm = m0 + wm * 16 + hi * 8;
    v8f accs[4] = {acc0, acc1, acc2, acc3};
    #pragma unroll
    for (int j = 0; j < 4; ++j) {
        const int cn = n0 + j * 16 + row;
        #pragma unroll
        for (int r = 0; r < 8; ++r) {
            size_t idx = (size_t)(cm + r) * N + cn;
            float v = accs[j][r];
            if (mode == 1)      v += res[idx];
            else if (mode == 2) v += res[cn];
            C[idx] = v;
        }
    }
}

// ---------------------------------------------------------------------------
// RMSNorm: one block per row
// ---------------------------------------------------------------------------
__global__ __launch_bounds__(256) void rmsnorm_kernel(
    const float* __restrict__ x, const float* __restrict__ g,
    float* __restrict__ y, int dim)
{
    const int b = blockIdx.x;
    const float* xr = x + (size_t)b * dim;
    float ss = 0.f;
    for (int i = threadIdx.x; i < dim; i += 256) { float v = xr[i]; ss += v * v; }
    __shared__ float red[256];
    red[threadIdx.x] = ss; __syncthreads();
    for (int s = 128; s > 0; s >>= 1) {
        if (threadIdx.x < s) red[threadIdx.x] += red[threadIdx.x + s];
        __syncthreads();
    }
    float inv = rsqrtf(red[0] / (float)dim + 1e-8f);
    float* yr = y + (size_t)b * dim;
    for (int i = threadIdx.x; i < dim; i += 256) yr[i] = xr[i] * g[i] * inv;
}

// ---------------------------------------------------------------------------
// RoPE (pos = s) on q,k; scatter k,v into caches at position s
// ---------------------------------------------------------------------------
__global__ void rope_cache_kernel(const float* __restrict__ qkv,
                                  float* __restrict__ qbuf,
                                  float* __restrict__ kcache,
                                  float* __restrict__ vcache,
                                  int s, int layer)
{
    int t = blockIdx.x * blockDim.x + threadIdx.x;     // b*128 + d
    if (t >= BATCH * INNER) return;
    int b = t >> 7, d = t & 127;
    const float* rowp = qkv + (size_t)b * 384;
    float qv = rowp[d], kv = rowp[128 + d], vv = rowp[256 + d];
    int hd = d & 63;
    if (hd < ROT_DIM) {
        float freq = powf(10000.f, -((float)(hd & 15)) / 16.f);
        float ang = (float)s * freq;
        float c = cosf(ang), sn = sinf(ang);
        if (hd < 16) {
            qv = qv * c - rowp[d + 16] * sn;
            kv = kv * c - rowp[128 + d + 16] * sn;
        } else {
            qv = qv * c + rowp[d - 16] * sn;
            kv = kv * c + rowp[128 + d - 16] * sn;
        }
    }
    qbuf[t] = qv;
    size_t cidx = ((((size_t)layer * SEQ_LEN + s) * BATCH) + b) * INNER + d;
    kcache[cidx] = kv;
    vcache[cidx] = vv;
}

// ---------------------------------------------------------------------------
// Attention for the current token: one thread per (b, head), seq <= 8
// ---------------------------------------------------------------------------
__global__ void attention_kernel(const float* __restrict__ qbuf,
                                 const float* __restrict__ kcache,
                                 const float* __restrict__ vcache,
                                 float* __restrict__ obuf, int s, int layer)
{
    int t = blockIdx.x * blockDim.x + threadIdx.x;     // b*HEADS + h
    if (t >= BATCH * HEADS) return;
    int b = t >> 1, hh = t & 1;
    const float* q = qbuf + (size_t)b * INNER + hh * DIM_HEAD;
    float att[SEQ_LEN];
    float m = -3.4e38f;
    for (int p = 0; p <= s; ++p) {
        const float* k = kcache + ((((size_t)layer * SEQ_LEN + p) * BATCH) + b) * INNER + hh * DIM_HEAD;
        float dot = 0.f;
        #pragma unroll 8
        for (int d = 0; d < DIM_HEAD; ++d) dot += q[d] * k[d];
        dot *= 0.125f;                                  // 1/sqrt(64)
        att[p] = dot;
        m = fmaxf(m, dot);
    }
    float denom = 0.f;
    for (int p = 0; p <= s; ++p) { att[p] = expf(att[p] - m); denom += att[p]; }
    float inv = 1.f / denom;
    float* o = obuf + (size_t)b * INNER + hh * DIM_HEAD;
    for (int d = 0; d < DIM_HEAD; ++d) {
        float a = 0.f;
        for (int p = 0; p <= s; ++p) {
            const float* v = vcache + ((((size_t)layer * SEQ_LEN + p) * BATCH) + b) * INNER + hh * DIM_HEAD;
            a += att[p] * v[d];
        }
        o[d] = a * inv;
    }
}

// ---------------------------------------------------------------------------
// GEGLU: out[b][j] = u[b][j] * gelu_tanh(u[b][2048+j])
// ---------------------------------------------------------------------------
__global__ void geglu_kernel(const float* __restrict__ u, float* __restrict__ out, int n) {
    int t = blockIdx.x * blockDim.x + threadIdx.x;
    if (t >= n) return;
    int b = t >> 11, j = t & 2047;
    float a = u[(size_t)b * FF2 + j];
    float g = u[(size_t)b * FF2 + FF_INNER + j];
    float gl = 0.5f * g * (1.f + tanhf(0.7978845608028654f * (g + 0.044715f * g * g * g)));
    out[t] = a * gl;
}

// ---------------------------------------------------------------------------
// Mask used classes with -inf
// ---------------------------------------------------------------------------
__global__ void mask_kernel(float* __restrict__ logits,
                            const unsigned char* __restrict__ used, int n) {
    int t = blockIdx.x * blockDim.x + threadIdx.x;
    if (t >= n) return;
    if (used[t]) logits[t] = -__builtin_inff();
}

// ---------------------------------------------------------------------------
// Row softmax -> probs (block per row)
// ---------------------------------------------------------------------------
__global__ __launch_bounds__(256) void softmax_rows_kernel(
    const float* __restrict__ logits, float* __restrict__ probs)
{
    const int b = blockIdx.x;
    const float* row = logits + (size_t)b * CLASSES;
    __shared__ float red[256];
    float m = -__builtin_inff();
    for (int c = threadIdx.x; c < CLASSES; c += 256) m = fmaxf(m, row[c]);
    red[threadIdx.x] = m; __syncthreads();
    for (int s = 128; s > 0; s >>= 1) {
        if (threadIdx.x < s) red[threadIdx.x] = fmaxf(red[threadIdx.x], red[threadIdx.x + s]);
        __syncthreads();
    }
    m = red[0]; __syncthreads();
    float ssum = 0.f;
    float* prow = probs + (size_t)b * CLASSES;
    for (int c = threadIdx.x; c < CLASSES; c += 256) {
        float e = expf(row[c] - m);
        prow[c] = e;
        ssum += e;
    }
    red[threadIdx.x] = ssum; __syncthreads();
    for (int s = 128; s > 0; s >>= 1) {
        if (threadIdx.x < s) red[threadIdx.x] += red[threadIdx.x + s];
        __syncthreads();
    }
    float inv = 1.f / red[0];
    for (int c = threadIdx.x; c < CLASSES; c += 256) prow[c] *= inv;
}

// Deterministic class mean over batch (fixed summation order)
__global__ void classmean_kernel(const float* __restrict__ probs, float* __restrict__ mp) {
    int c = blockIdx.x * blockDim.x + threadIdx.x;
    if (c >= CLASSES) return;
    float s = 0.f;
    for (int b = 0; b < BATCH; ++b) s += probs[(size_t)b * CLASSES + c];
    mp[c] = s / (float)BATCH;
}

__global__ __launch_bounds__(256) void perp_kernel(const float* __restrict__ mp,
                                                   float* __restrict__ perp_out, int s) {
    __shared__ float red[256];
    float acc = 0.f;
    for (int c = threadIdx.x; c < CLASSES; c += 256) {
        float m = mp[c];
        acc += m * logf(m + 1e-7f);
    }
    red[threadIdx.x] = acc; __syncthreads();
    for (int st = 128; st > 0; st >>= 1) {
        if (threadIdx.x < st) red[threadIdx.x] += red[threadIdx.x + st];
        __syncthreads();
    }
    if (threadIdx.x == 0) {
        float p = expf(-red[0]) / (float)SEQ_LEN;
        if (s == 0) *perp_out = p; else *perp_out += p;
    }
}

// ---------------------------------------------------------------------------
// Gumbel sampling + argmax (block per row, deterministic hash noise)
// ---------------------------------------------------------------------------
__device__ __forceinline__ unsigned int hash3(unsigned int a, unsigned int b, unsigned int c) {
    unsigned int h = a * 0x9E3779B1u ^ b * 0x85EBCA77u ^ c * 0xC2B2AE3Du;
    h ^= h >> 16; h *= 0x7FEB352Du; h ^= h >> 15; h *= 0x846CA68Bu; h ^= h >> 16;
    return h;
}

__global__ __launch_bounds__(256) void sample_kernel(
    const float* __restrict__ logits, unsigned char* __restrict__ used,
    int* __restrict__ idxbuf, float* __restrict__ codes_out, int s)
{
    const int b = blockIdx.x;
    const float* row = logits + (size_t)b * CLASSES;
    float best = -__builtin_inff();
    int bidx = CLASSES;
    for (int c = threadIdx.x; c < CLASSES; c += 256) {
        unsigned int h = hash3((unsigned)s, (unsigned)b, (unsigned)c);
        float u = (float)(h >> 8) * (1.0f / 16777216.0f) + 1e-10f;
        float g = -logf(-logf(u));
        float v = row[c] + g;                 // TAU == 1
        if (v > best || (v == best && c < bidx)) { best = v; bidx = c; }
    }
    __shared__ float bv[256];
    __shared__ int   bi[256];
    bv[threadIdx.x] = best; bi[threadIdx.x] = bidx; __syncthreads();
    for (int st = 128; st > 0; st >>= 1) {
        if (threadIdx.x < st) {
            float ov = bv[threadIdx.x + st]; int oi = bi[threadIdx.x + st];
            if (ov > bv[threadIdx.x] || (ov == bv[threadIdx.x] && oi < bi[threadIdx.x])) {
                bv[threadIdx.x] = ov; bi[threadIdx.x] = oi;
            }
        }
        __syncthreads();
    }
    if (threadIdx.x == 0) {
        int idx = bi[0];
        if (idx >= CLASSES) idx = 0;
        idxbuf[b] = idx;
        used[(size_t)b * CLASSES + idx] = 1;
        codes_out[(size_t)s * BATCH + b] = (float)idx;
    }
}

// ---------------------------------------------------------------------------
// One-hot feedback gather + cumulative output  (c == y_hard)
// ---------------------------------------------------------------------------
__global__ __launch_bounds__(256) void feedback_kernel(
    const int* __restrict__ idxbuf, const float* __restrict__ fb_w,
    const float* __restrict__ cb_w, float* __restrict__ h_next,
    float* __restrict__ accum, float* __restrict__ results, int s)
{
    const int b = blockIdx.x;
    const int idx = idxbuf[b];
    for (int d = threadIdx.x; d < DIM; d += 256) {
        h_next[(size_t)b * DIM + d] = fb_w[(size_t)d * CLASSES + idx];
        float cv = (idx == 0) ? 0.f : cb_w[(size_t)d * CLASSES + idx];   // col 0 zeroed
        float a = (s == 0) ? cv : (accum[(size_t)b * OUT_DIM + d] + cv);
        accum[(size_t)b * OUT_DIM + d] = a;
        results[((size_t)s * BATCH + b) * OUT_DIM + d] = a;
    }
}

// ===========================================================================
extern "C" void kernel_launch(void* const* d_in, const int* in_sizes, int n_in,
                              void* d_out, int out_size, void* d_ws, size_t ws_size,
                              hipStream_t stream)
{
    const float* x       = (const float*)d_in[0];
    const float* rms1_g  = (const float*)d_in[1];
    const float* wq      = (const float*)d_in[2];
    const float* wk      = (const float*)d_in[3];
    const float* wv      = (const float*)d_in[4];
    const float* wo      = (const float*)d_in[5];
    const float* rms2_g  = (const float*)d_in[6];
    const float* ff_w1   = (const float*)d_in[7];
    const float* ff_w2   = (const float*)d_in[8];
    const float* final_g = (const float*)d_in[9];
    const float* cls_w   = (const float*)d_in[10];
    const float* cls_b   = (const float*)d_in[11];
    const float* fb_w    = (const float*)d_in[12];
    const float* cb_w    = (const float*)d_in[13];

    float* out_f     = (float*)d_out;
    float* results   = out_f;                                  // [8][2048][1024]
    float* perp_out  = out_f + (size_t)SEQ_LEN * BATCH * OUT_DIM;
    float* codes_out = perp_out + 1;                           // [8*2048]

    // -------- workspace carve --------
    float* W = (float*)d_ws;
    size_t o = 0;
    auto alloc = [&](size_t n) { float* p = W + o; o += n; return p; };
    float* h      = alloc((size_t)BATCH * DIM);
    float* nbuf   = alloc((size_t)BATCH * DIM);
    float* qkvbuf = alloc((size_t)BATCH * 384);
    float* qbuf   = alloc((size_t)BATCH * INNER);
    float* obuf   = alloc((size_t)BATCH * INNER);
    float* ubuf   = alloc((size_t)BATCH * FF2);
    float* gbuf   = alloc((size_t)BATCH * FF_INNER);
    float* logits = alloc((size_t)BATCH * CLASSES);
    float* probs  = alloc((size_t)BATCH * CLASSES);
    float* mp     = alloc(CLASSES);
    float* accum  = alloc((size_t)BATCH * OUT_DIM);
    float* kcache = alloc((size_t)DEPTH * SEQ_LEN * BATCH * INNER);
    float* vcache = alloc((size_t)DEPTH * SEQ_LEN * BATCH * INNER);
    int*   idxbuf = (int*)alloc(BATCH);
    unsigned char* used = (unsigned char*)alloc(((size_t)BATCH * CLASSES) / 4);
    __bf16* qkvT  = (__bf16*)(W + o);  o += ((size_t)DEPTH * 384 * DIM) / 2;
    __bf16* woT   = (__bf16*)(W + o);  o += ((size_t)DEPTH * DIM * INNER) / 2;
    __bf16* ffw1T = (__bf16*)(W + o);  o += ((size_t)DEPTH * FF2 * DIM) / 2;
    __bf16* ffw2T = (__bf16*)(W + o);  o += ((size_t)DEPTH * DIM * FF_INNER) / 2;
    __bf16* clsT  = (__bf16*)(W + o);  o += ((size_t)CLASSES * DIM) / 2;

    const int T = 256;
    #define NB(n) ((int)(((size_t)(n) + T - 1) / T))

    // -------- weight conversion (every call; deterministic) --------
    for (int l = 0; l < DEPTH; ++l) {
        size_t nqkv = (size_t)DIM * INNER;
        convert_transpose_kernel<<<NB(nqkv), T, 0, stream>>>(
            wq + (size_t)l * nqkv, qkvT + (size_t)l * 384 * DIM, DIM, INNER, 0, DIM);
        convert_transpose_kernel<<<NB(nqkv), T, 0, stream>>>(
            wk + (size_t)l * nqkv, qkvT + (size_t)l * 384 * DIM, DIM, INNER, 128, DIM);
        convert_transpose_kernel<<<NB(nqkv), T, 0, stream>>>(
            wv + (size_t)l * nqkv, qkvT + (size_t)l * 384 * DIM, DIM, INNER, 256, DIM);
        convert_transpose_kernel<<<NB((size_t)INNER * DIM), T, 0, stream>>>(
            wo + (size_t)l * INNER * DIM, woT + (size_t)l * DIM * INNER, INNER, DIM, 0, INNER);
        convert_transpose_kernel<<<NB((size_t)DIM * FF2), T, 0, stream>>>(
            ff_w1 + (size_t)l * DIM * FF2, ffw1T + (size_t)l * FF2 * DIM, DIM, FF2, 0, DIM);
        convert_transpose_kernel<<<NB((size_t)FF_INNER * DIM), T, 0, stream>>>(
            ff_w2 + (size_t)l * FF_INNER * DIM, ffw2T + (size_t)l * DIM * FF_INNER, FF_INNER, DIM, 0, FF_INNER);
    }
    convert_kernel<<<NB((size_t)CLASSES * DIM), T, 0, stream>>>(cls_w, clsT, (size_t)CLASSES * DIM);

    // -------- per-launch state init --------
    zero_u32_kernel<<<NB(((size_t)BATCH * CLASSES) / 4), T, 0, stream>>>(
        (unsigned int*)used, ((size_t)BATCH * CLASSES) / 4);
    copy_kernel<<<NB((size_t)BATCH * DIM), T, 0, stream>>>(x, h, (size_t)BATCH * DIM);

    const dim3 gQKV(384 / 128, BATCH / 64);
    const dim3 gDIM(DIM / 128, BATCH / 64);
    const dim3 gFF2(FF2 / 128, BATCH / 64);
    const dim3 gCLS(CLASSES / 128, BATCH / 64);

    for (int s = 0; s < SEQ_LEN; ++s) {
        for (int l = 0; l < DEPTH; ++l) {
            rmsnorm_kernel<<<BATCH, T, 0, stream>>>(h, rms1_g + (size_t)l * DIM, nbuf, DIM);
            gemm_bf16_wmma<<<gQKV, T, 0, stream>>>(
                nbuf, qkvT + (size_t)l * 384 * DIM, qkvbuf, nullptr, BATCH, 384, DIM, 0);
            rope_cache_kernel<<<NB(BATCH * INNER), T, 0, stream>>>(
                qkvbuf, qbuf, kcache, vcache, s, l);
            attention_kernel<<<NB(BATCH * HEADS), T, 0, stream>>>(
                qbuf, kcache, vcache, obuf, s, l);
            gemm_bf16_wmma<<<gDIM, T, 0, stream>>>(
                obuf, woT + (size_t)l * DIM * INNER, h, h, BATCH, DIM, INNER, 1);
            rmsnorm_kernel<<<BATCH, T, 0, stream>>>(h, rms2_g + (size_t)l * DIM, nbuf, DIM);
            gemm_bf16_wmma<<<gFF2, T, 0, stream>>>(
                nbuf, ffw1T + (size_t)l * FF2 * DIM, ubuf, nullptr, BATCH, FF2, DIM, 0);
            geglu_kernel<<<NB(BATCH * FF_INNER), T, 0, stream>>>(ubuf, gbuf, BATCH * FF_INNER);
            gemm_bf16_wmma<<<gDIM, T, 0, stream>>>(
                gbuf, ffw2T + (size_t)l * DIM * FF_INNER, h, h, BATCH, DIM, FF_INNER, 1);
        }
        rmsnorm_kernel<<<BATCH, T, 0, stream>>>(h, final_g, nbuf, DIM);
        gemm_bf16_wmma<<<gCLS, T, 0, stream>>>(
            nbuf, clsT, logits, cls_b, BATCH, CLASSES, DIM, 2);
        mask_kernel<<<NB(BATCH * CLASSES), T, 0, stream>>>(logits, used, BATCH * CLASSES);
        softmax_rows_kernel<<<BATCH, T, 0, stream>>>(logits, probs);
        classmean_kernel<<<CLASSES / T, T, 0, stream>>>(probs, mp);
        perp_kernel<<<1, T, 0, stream>>>(mp, perp_out, s);
        sample_kernel<<<BATCH, T, 0, stream>>>(logits, used, idxbuf, codes_out, s);
        feedback_kernel<<<BATCH, T, 0, stream>>>(idxbuf, fb_w, cb_w, h, accum, results, s);
    }
    #undef NB
}
